// VectorQuantizer_5609227288908
// MI455X (gfx1250) — matrix operand order repaired and checked by reading, not compiled, not run on hardware
//
#include <hip/hip_runtime.h>
#include <stdint.h>

typedef __attribute__((ext_vector_type(16))) _Float16 v16h;
typedef __attribute__((ext_vector_type(8)))  float    v8f;

#define C_DIM  64
#define K_DIM  4096
#define TILE_M 16

// Deterministic hash -> Gumbel(0,1) noise (stand-in for jax.random.gumbel(key(1)))
__device__ __forceinline__ float gumbel_from_hash(uint32_t x) {
    x ^= x >> 17; x *= 0xed5ad4bbu;
    x ^= x >> 11; x *= 0xac4c1b51u;
    x ^= x >> 15; x *= 0x31848babu;
    x ^= x >> 14;
    float u = (float)(x >> 8) * (1.0f / 16777216.0f) + 5.9604645e-8f; // (0,1)
    return -__logf(-__logf(u));
}

// Prep: embeddings f32 -> f16 copy (L2-resident operand for WMMA) + ||e||^2
__global__ void vq_prep_kernel(const float* __restrict__ emb,
                               _Float16* __restrict__ emb16,
                               float* __restrict__ enorm, int K) {
    int k = blockIdx.x * blockDim.x + threadIdx.x;
    if (k >= K) return;
    const float* row  = emb   + (size_t)k * C_DIM;
    _Float16*    drow = emb16 + (size_t)k * C_DIM;
    float s = 0.f;
#pragma unroll
    for (int c = 0; c < C_DIM; c += 4) {
        float4 v = *(const float4*)(row + c);
        s += v.x * v.x + v.y * v.y + v.z * v.z + v.w * v.w;
        drow[c + 0] = (_Float16)v.x;
        drow[c + 1] = (_Float16)v.y;
        drow[c + 2] = (_Float16)v.z;
        drow[c + 3] = (_Float16)v.w;
    }
    enorm[k] = s;
}

// Fused: WMMA logits (16 rows x 4096 codes in LDS) -> gumbel argmax ->
// log_softmax -> single coalesced HBM write + codebook gather.
__global__ void __launch_bounds__(256)
vq_fused_kernel(const float* __restrict__ keys,
                const float* __restrict__ emb,        // f32 [K][64] (for gather)
                const _Float16* __restrict__ emb16,   // f16 [K][64]
                const float* __restrict__ enormG,     // [K]
                float* __restrict__ samples,          // [N][64]
                float* __restrict__ logp)             // [N][K]
{
    extern __shared__ float smem[];
    float* logitsS = smem;                          // 16 * 4096 f32 = 256 KB
    float* knormS  = logitsS + TILE_M * K_DIM;      // 16
    float* redMax  = knormS + 16;                   // 256
    float* redPer  = redMax + 256;                  // 256
    float* redSum  = redPer + 256;                  // 256
    float* rowMax  = redSum + 256;                  // 16
    float* logZ    = rowMax + 16;                   // 16
    int*   redIdx  = (int*)(logZ + 16);             // 256
    int*   rowIdx  = redIdx + 256;                  // 16

    const int tid     = threadIdx.x;
    const int lane    = tid & 31;
    const int wave    = tid >> 5;
    const int rowBase = blockIdx.x * TILE_M;

    // ---- ||keys_row||^2 for the 16 rows of this tile ----
    if (tid < TILE_M) {
        const float* krow = keys + (size_t)(rowBase + tid) * C_DIM;
        float s = 0.f;
#pragma unroll
        for (int c = 0; c < C_DIM; c += 4) {
            float4 v = *(const float4*)(krow + c);
            s += v.x * v.x + v.y * v.y + v.z * v.z + v.w * v.w;
        }
        knormS[tid] = s;
    }
    __syncthreads();

    // ---- A fragments: keys tile, f32 -> f16, contiguous-k 16-bit A 16x32 layout ----
    // lane l (0-15): M=l, halves 0-15 = K h*16 .. h*16+15 (contiguous, matches B)
    const int nn = lane & 15;
    const int h  = lane >> 4;
    v16h a0, a1;
    {
        const float* krow = keys + (size_t)(rowBase + nn) * C_DIM;
#pragma unroll
        for (int j = 0; j < 8; ++j) {
            a0[j]     = (_Float16)krow[h * 16 + j];
            a0[j + 8] = (_Float16)krow[h * 16 + 8 + j];
            a1[j]     = (_Float16)krow[32 + h * 16 + j];
            a1[j + 8] = (_Float16)krow[32 + h * 16 + 8 + j];
        }
    }

    // ---- logits: each wave computes 32 N-tiles of 16 codes ----
    for (int t = 0; t < 32; ++t) {
        const int cb = (((t << 3) | wave) << 4);      // code tile base, waves interleaved
        const int n  = cb + nn;                       // this lane's code column
        // B 32x16 f16 layout: lane holds 16 contiguous halves of code row n
        const _Float16* brow = emb16 + (size_t)n * C_DIM + h * 16;
        v16h b0 = *(const v16h*)(brow);               // channels [h*16 .. h*16+15]
        v16h b1 = *(const v16h*)(brow + 32);          // channels [32+h*16 ..]
        v8f acc = {};
        acc = __builtin_amdgcn_wmma_f32_16x16x32_f16(false, a0, false, b0,
                                                     (short)0, acc, false, false);
        acc = __builtin_amdgcn_wmma_f32_16x16x32_f16(false, a1, false, b1,
                                                     (short)0, acc, false, false);
        const float en = enormG[n];
#pragma unroll
        for (int j = 0; j < 8; ++j) {
            const int m = j + 8 * h;                  // C/D layout: VGPR j -> M=j / j+8
            logitsS[m * K_DIM + n] = 2.0f * acc[j] - knormS[m] - en;
        }
    }
    __syncthreads();

    // ---- per-row reductions: 16 threads per row, strided cols for bank spread ----
    const int r = tid >> 4;                           // row in tile
    const int g = tid & 15;                           // scan group
    const float* lrow = logitsS + r * K_DIM;
    const uint32_t hbase = (uint32_t)(rowBase + r) * (uint32_t)K_DIM;

    float mx = -3.4e38f, pmx = -3.4e38f;
    int pidx = 0;
    for (int i = 0; i < 256; ++i) {
        const int col = g + (i << 4);
        const float L = lrow[col];
        mx = fmaxf(mx, L);
        const float p = L + gumbel_from_hash(hbase + (uint32_t)col);
        if (p > pmx) { pmx = p; pidx = col; }
    }
    redMax[tid] = mx; redPer[tid] = pmx; redIdx[tid] = pidx;
    __syncthreads();
    if (tid < TILE_M) {
        float m2 = -3.4e38f, p2 = -3.4e38f; int i2 = 0;
#pragma unroll
        for (int q = 0; q < 16; ++q) {
            const int s = tid * 16 + q;
            m2 = fmaxf(m2, redMax[s]);
            if (redPer[s] > p2) { p2 = redPer[s]; i2 = redIdx[s]; }
        }
        rowMax[tid] = m2; rowIdx[tid] = i2;
    }
    __syncthreads();

    const float rm = rowMax[r];
    float se = 0.f;
    for (int i = 0; i < 256; ++i)
        se += __expf(lrow[g + (i << 4)] - rm);
    redSum[tid] = se;
    __syncthreads();
    if (tid < TILE_M) {
        float s = 0.f;
#pragma unroll
        for (int q = 0; q < 16; ++q) s += redSum[tid * 16 + q];
        logZ[tid] = rowMax[tid] + __logf(s);
    }
    __syncthreads();

    // ---- single coalesced HBM write of log_probs (float4 per lane) ----
    for (int rr = 0; rr < TILE_M; ++rr) {
        const float lz = logZ[rr];
        const float* srow = logitsS + rr * K_DIM;
        float* orow = logp + (size_t)(rowBase + rr) * K_DIM;
#pragma unroll
        for (int it = 0; it < 4; ++it) {
            const int col = it * 1024 + tid * 4;
            float4 v = *(const float4*)(srow + col);
            v.x -= lz; v.y -= lz; v.z -= lz; v.w -= lz;
            *(float4*)(orow + col) = v;
        }
    }

    // ---- samples = embeddings[argmax] (straight-through forward) ----
    {
        const int idx = rowIdx[r];
        const float4 v = *(const float4*)(emb + (size_t)idx * C_DIM + g * 4);
        *(float4*)(samples + (size_t)(rowBase + r) * C_DIM + g * 4) = v;
    }
}

extern "C" void kernel_launch(void* const* d_in, const int* in_sizes, int n_in,
                              void* d_out, int out_size, void* d_ws, size_t ws_size,
                              hipStream_t stream) {
    const float* keys = (const float*)d_in[0];
    const float* emb  = (const float*)d_in[1];
    const int N = in_sizes[0] / C_DIM;   // 65536
    const int K = in_sizes[1] / C_DIM;   // 4096

    _Float16* emb16 = (_Float16*)d_ws;
    float* enorm = (float*)((char*)d_ws + (size_t)K * C_DIM * sizeof(_Float16));

    float* out     = (float*)d_out;
    float* samples = out;                         // [N][64]
    float* logp    = out + (size_t)N * C_DIM;     // [N][K]

    vq_prep_kernel<<<(K + 255) / 256, 256, 0, stream>>>(emb, emb16, enorm, K);

    const size_t smemBytes =
        (size_t)(TILE_M * K_DIM + 16 + 3 * 256 + 2 * 16) * sizeof(float) +
        (size_t)(256 + 16) * sizeof(int);         // ~266 KB < 320 KB/WGP
    vq_fused_kernel<<<N / TILE_M, 256, smemBytes, stream>>>(
        keys, emb, emb16, enorm, samples, logp);
}
